// Decoder_12060268167529
// MI455X (gfx1250) — matrix operand order repaired and checked by reading, not compiled, not run on hardware
//
#include <hip/hip_runtime.h>

typedef __attribute__((ext_vector_type(16))) __bf16 v16bf;
typedef __attribute__((ext_vector_type(8)))  float  v8f;
typedef __attribute__((ext_vector_type(4)))  unsigned int u32x4;
typedef __attribute__((ext_vector_type(4)))  float  f32x4;

#define TT   2048
#define BB   4
#define DD   512
#define HH   8
#define VV   256
#define LLAY 8
#define FFD  2048
#define MTOK (BB * TT)   // 8192 tokens

// ---------- helpers ----------
static __device__ __forceinline__ __bf16 f2bf(float f) {
  unsigned u = __float_as_uint(f);
  u += 0x7FFFu + ((u >> 16) & 1u);                 // round-to-nearest-even
  unsigned short s = (unsigned short)(u >> 16);
  return __builtin_bit_cast(__bf16, s);
}

struct U32x8  { u32x4 a, b; };
struct B16x16 { __bf16 e[16]; };

static __device__ __forceinline__ v16bf ld_frag(const __bf16* p0, const __bf16* p1) {
  U32x8 t;
  t.a = *(const u32x4*)p0;
  t.b = *(const u32x4*)p1;
  return __builtin_bit_cast(v16bf, t);
}

static __device__ __forceinline__ v8f zero8() {
  v8f z;
#pragma unroll
  for (int i = 0; i < 8; ++i) z[i] = 0.f;
  return z;
}

static __device__ __forceinline__ v8f wmma_bf16(v16bf a, v16bf b, v8f c) {
  return __builtin_amdgcn_wmma_f32_16x16x32_bf16(false, a, false, b, (short)0, c, false, false);
}

// async copy 16B: global -> LDS (CDNA5 path, bypasses VGPRs, tracked by ASYNCcnt)
static __device__ __forceinline__ void async_cp16(unsigned lds_off, const void* gptr) {
  asm volatile("global_load_async_to_lds_b128 %0, %1, off"
               :: "v"(lds_off), "v"((unsigned long long)gptr) : "memory");
}
static __device__ __forceinline__ void wait_async_le4() {
  asm volatile("s_wait_asynccnt 0x4" ::: "memory");
}
static __device__ __forceinline__ void wait_async_0() {
  asm volatile("s_wait_asynccnt 0x0" ::: "memory");
}

// ---------- weight pre-pass: straight f32 -> bf16 ----------
__global__ __launch_bounds__(256)
void cvt_kernel(const float* __restrict__ in, __bf16* __restrict__ out) {
  int i = blockIdx.x * 256 + threadIdx.x;
  out[i] = f2bf(in[i]);
}

// ---------- weight pre-pass: f32 [K,N] -> bf16 [N,K] (LDS 32x32 tile transpose) ----------
__global__ __launch_bounds__(256)
void tcvt_kernel(const float* __restrict__ in, __bf16* __restrict__ out, int K, int N) {
  __shared__ float S[32][33];
  int n0 = blockIdx.x * 32, k0 = blockIdx.y * 32;
  int c = threadIdx.x & 31, rr = threadIdx.x >> 5;
#pragma unroll
  for (int e = 0; e < 4; ++e)
    S[rr + e * 8][c] = in[(size_t)(k0 + rr + e * 8) * N + n0 + c];
  __syncthreads();
#pragma unroll
  for (int e = 0; e < 4; ++e)
    out[(size_t)(n0 + rr + e * 8) * K + k0 + c] = f2bf(S[c][rr + e * 8]);
}

// ---------- embedding: h = tok_emb[x] + pos_emb ----------
__global__ __launch_bounds__(256)
void embed_kernel(const int* __restrict__ x, const float* __restrict__ tok,
                  const float* __restrict__ pos, float* __restrict__ h) {
  int idx = blockIdx.x * 256 + threadIdx.x;        // over MTOK*DD
  int bt  = idx >> 9;                              // /512
  int d   = idx & 511;
  int t   = bt & (TT - 1);
  int v   = x[bt];
  h[idx]  = tok[v * DD + d] + pos[t * DD + d];
}

// ---------- LayerNorm: one wave per token, vectorized b128 I/O, bf16 out ----------
__global__ __launch_bounds__(256)
void ln_kernel(const float* __restrict__ x, const float* __restrict__ g,
               const float* __restrict__ b, __bf16* __restrict__ out) {
  int w = threadIdx.x >> 5, lane = threadIdx.x & 31;
  size_t row = (size_t)blockIdx.x * 8 + w;
  const f32x4* xr = (const f32x4*)(x + row * DD) + lane * 4;  // 16 floats / lane
  f32x4 v[4]; float s = 0.f;
#pragma unroll
  for (int e = 0; e < 4; ++e) {
    v[e] = xr[e];
#pragma unroll
    for (int c = 0; c < 4; ++c) s += v[e][c];
  }
#pragma unroll
  for (int m = 16; m >= 1; m >>= 1) s += __shfl_xor(s, m, 32);
  float mean = s * (1.f / DD);
  float q = 0.f;
#pragma unroll
  for (int e = 0; e < 4; ++e)
#pragma unroll
    for (int c = 0; c < 4; ++c) { float d = v[e][c] - mean; q += d * d; }
#pragma unroll
  for (int m = 16; m >= 1; m >>= 1) q += __shfl_xor(q, m, 32);
  float rstd = rsqrtf(q * (1.f / DD) + 1e-5f);
  const f32x4* gp = (const f32x4*)g + lane * 4;
  const f32x4* bp = (const f32x4*)b + lane * 4;
  B16x16 ob;
#pragma unroll
  for (int e = 0; e < 4; ++e) {
    f32x4 gv = gp[e], bv = bp[e];
#pragma unroll
    for (int c = 0; c < 4; ++c)
      ob.e[e * 4 + c] = f2bf((v[e][c] - mean) * rstd * gv[c] + bv[c]);
  }
  U32x8 uu = __builtin_bit_cast(U32x8, ob);
  __bf16* orow = out + row * DD + lane * 16;
  *(u32x4*)orow       = uu.a;
  *(u32x4*)(orow + 8) = uu.b;
}

// ---------- WMMA GEMM: C[M,N] = A_bf16[M,K] * Bt_bf16[N,K]^T (+bias,gelu,resid) ----------
// block 256 = 8 waves as 2(M)x4(N); tile 128x128, BK=32; each wave 64x32 (4x2 accums).
// Double-buffered LDS fed by global_load_async_to_lds_b128 (4 async issues/wave/tile).
// SCALEQ: multiply columns [0,512) by 1/sqrt(HD) (folds attention score scale into Q).
template<bool BIAS, bool GELU, bool RESID, bool OUT_BF16, bool OUT_F32, bool SCALEQ>
__global__ __launch_bounds__(256)
void gemm_kernel(const __bf16* __restrict__ A, const __bf16* __restrict__ Bt,
                 const float* __restrict__ bias, const float* resid,
                 float* outF, __bf16* outB, int N, int K) {
  __shared__ __bf16 A_lds[2][128][40];   // [buf][m][k], padded
  __shared__ __bf16 B_lds[2][128][40];   // [buf][n][k], padded
  const int tid = threadIdx.x;
  const int w = tid >> 5, lane = tid & 31;
  const int r = lane & 15, hi = lane >> 4;
  const int wm = w >> 2, wn = w & 3;
  const int m0 = blockIdx.y * 128;
  const int n0 = blockIdx.x * 128;

  // loader mapping: each thread owns 16 bf16 (2x16B) of the A tile and of the B tile
  const int lrow = tid >> 1, lhalf = tid & 1;
  const __bf16* aG = A  + (size_t)(m0 + lrow) * K + lhalf * 16;
  const __bf16* bG = Bt + (size_t)(n0 + lrow) * K + lhalf * 16;

  v8f acc[4][2];
#pragma unroll
  for (int mi = 0; mi < 4; ++mi)
#pragma unroll
    for (int ni = 0; ni < 2; ++ni) acc[mi][ni] = zero8();

  auto prefetch = [&](int kt, int buf) {
    unsigned la = (unsigned)(unsigned long long)&A_lds[buf][lrow][lhalf * 16];
    unsigned lb = (unsigned)(unsigned long long)&B_lds[buf][lrow][lhalf * 16];
    async_cp16(la,      aG + kt);
    async_cp16(la + 16, aG + kt + 8);
    async_cp16(lb,      bG + kt);
    async_cp16(lb + 16, bG + kt + 8);
  };

  const int nt = K >> 5;
  prefetch(0, 0);
  for (int t = 0; t < nt; ++t) {
    const int cur = t & 1;
    if (t + 1 < nt) { prefetch((t + 1) << 5, cur ^ 1); wait_async_le4(); }
    else            { wait_async_0(); }
    __syncthreads();

    v16bf bf[2];
#pragma unroll
    for (int ni = 0; ni < 2; ++ni) {   // B frag: chunks at K = hi*8 and 16+hi*8
      const __bf16* p = &B_lds[cur][wn * 32 + ni * 16 + r][0];
      bf[ni] = ld_frag(p + hi * 8, p + 16 + hi * 8);
    }
#pragma unroll
    for (int mi = 0; mi < 4; ++mi) {   // A frag: chunks at K = hi*8 and 16+hi*8
      const __bf16* p = &A_lds[cur][wm * 64 + mi * 16 + r][0];
      v16bf af = ld_frag(p + hi * 8, p + 16 + hi * 8);
#pragma unroll
      for (int ni = 0; ni < 2; ++ni) acc[mi][ni] = wmma_bf16(af, bf[ni], acc[mi][ni]);
    }
    __syncthreads();
  }
  // epilogue: C layout row = i + hi*8, col = lane&15
#pragma unroll
  for (int mi = 0; mi < 4; ++mi)
#pragma unroll
    for (int ni = 0; ni < 2; ++ni) {
      int gn = n0 + wn * 32 + ni * 16 + r;
      float bv = 0.f;
      if constexpr (BIAS) bv = bias[gn];
      float sc = 1.f;
      if constexpr (SCALEQ) sc = (gn < DD) ? 0.125f : 1.f;   // 1/sqrt(64) on Q columns
#pragma unroll
      for (int i = 0; i < 8; ++i) {
        int gm = m0 + wm * 64 + mi * 16 + i + hi * 8;
        float v = acc[mi][ni][i] + bv;
        if constexpr (GELU) v = 0.5f * v * (1.f + erff(v * 0.70710678118f));
        if constexpr (SCALEQ) v *= sc;
        size_t o = (size_t)gm * N + gn;
        if constexpr (RESID)    v += resid[o];
        if constexpr (OUT_F32)  outF[o] = v;
        if constexpr (OUT_BF16) outB[o] = f2bf(v);
      }
    }
}

// ---------- flash attention: block = 4 waves x 16 queries (BQ=64), BK=32 ----------
// Q pre-scaled by 1/sqrt(HD) in the QKV GEMM epilogue.
__global__ __launch_bounds__(128)
void attn_kernel(const __bf16* __restrict__ qkv, __bf16* __restrict__ obuf) {
  __shared__ __bf16 K_lds[32][72];      // [key][d]
  __shared__ __bf16 VT_lds[64][40];     // [d][key]
  __shared__ __bf16 P_lds[4][16][40];   // per-wave P transpose staging
  const int bh = blockIdx.y;
  const int b = bh >> 3, h = bh & 7;
  const int qbase = blockIdx.x * 64;
  const int w = threadIdx.x >> 5, lane = threadIdx.x & 31;
  const int r = lane & 15, hi = lane >> 4;

  // preload Q fragments (HD=64 -> 2 K-steps of 32)
  const int qrow = qbase + w * 16 + r;
  const __bf16* qp = qkv + (size_t)(b * TT + qrow) * 1536 + h * 64;
  v16bf qf[2];
#pragma unroll
  for (int f = 0; f < 2; ++f)
    qf[f] = ld_frag(qp + f * 32 + hi * 8, qp + f * 32 + 16 + hi * 8);

  float ms[8], ls[8];
#pragma unroll
  for (int i = 0; i < 8; ++i) { ms[i] = -1e30f; ls[i] = 0.f; }
  v8f o[4];
#pragma unroll
  for (int n = 0; n < 4; ++n) o[n] = zero8();

  const int tkey = threadIdx.x >> 2, tseg = threadIdx.x & 3;
  for (int kt = 0; kt < qbase + 64; kt += 32) {
    __syncthreads();
    { // cooperative K/V tile load (32 keys x 64 d), V stored transposed
      const __bf16* kp = qkv + (size_t)(b * TT + kt + tkey) * 1536 + 512 + h * 64 + tseg * 16;
      *(u32x4*)&K_lds[tkey][tseg * 16]     = *(const u32x4*)kp;
      *(u32x4*)&K_lds[tkey][tseg * 16 + 8] = *(const u32x4*)(kp + 8);
      const __bf16* vp = qkv + (size_t)(b * TT + kt + tkey) * 1536 + 1024 + h * 64 + tseg * 16;
      U32x8 uu; uu.a = *(const u32x4*)vp; uu.b = *(const u32x4*)(vp + 8);
      B16x16 vv = __builtin_bit_cast(B16x16, uu);
#pragma unroll
      for (int e = 0; e < 16; ++e) VT_lds[tseg * 16 + e][tkey] = vv.e[e];
    }
    __syncthreads();
    // S = Q * K^T  (B frag read contiguous from row-major K tile)
    v8f s[2] = { zero8(), zero8() };
#pragma unroll
    for (int kk = 0; kk < 2; ++kk)
#pragma unroll
      for (int ni = 0; ni < 2; ++ni) {
        const __bf16* p = &K_lds[ni * 16 + r][kk * 32];
        s[ni] = wmma_bf16(qf[kk], ld_frag(p + hi * 8, p + 16 + hi * 8), s[ni]);
      }
    // causal mask only on diagonal-crossing tiles (wave-uniform branch)
    if (kt + 31 > qbase + w * 16) {
#pragma unroll
      for (int ni = 0; ni < 2; ++ni) {
        int kg = kt + ni * 16 + r;
#pragma unroll
        for (int i = 0; i < 8; ++i) {
          int qg = qbase + w * 16 + i + hi * 8;
          if (kg > qg) s[ni][i] = -1e30f;
        }
      }
    }
    // online softmax (row stats; butterfly stays in 16-lane half)
#pragma unroll
    for (int i = 0; i < 8; ++i) {
      float tm = fmaxf(s[0][i], s[1][i]);
#pragma unroll
      for (int m = 8; m >= 1; m >>= 1) tm = fmaxf(tm, __shfl_xor(tm, m, 32));
      float mn = fmaxf(ms[i], tm);
      float alpha = __expf(ms[i] - mn);
      float p0 = __expf(s[0][i] - mn);
      float p1 = __expf(s[1][i] - mn);
      s[0][i] = p0; s[1][i] = p1;
      float rs = p0 + p1;
#pragma unroll
      for (int m = 8; m >= 1; m >>= 1) rs += __shfl_xor(rs, m, 32);
      ls[i] = ls[i] * alpha + rs;
      ms[i] = mn;
#pragma unroll
      for (int n = 0; n < 4; ++n) o[n][i] *= alpha;
    }
    // transpose P through per-wave LDS (LDS ops in-order per wave)
#pragma unroll
    for (int ni = 0; ni < 2; ++ni)
#pragma unroll
      for (int i = 0; i < 8; ++i)
        P_lds[w][i + hi * 8][ni * 16 + r] = f2bf(s[ni][i]);
    v16bf pa = ld_frag(&P_lds[w][r][hi * 8], &P_lds[w][r][16 + hi * 8]);
    // O += P * V  (B frag contiguous from transposed V tile)
#pragma unroll
    for (int n = 0; n < 4; ++n) {
      const __bf16* p = &VT_lds[n * 16 + r][0];
      o[n] = wmma_bf16(pa, ld_frag(p + hi * 8, p + 16 + hi * 8), o[n]);
    }
  }
  // normalize and store bf16 (proj GEMM input), layout [B,T, h*64+d]
#pragma unroll
  for (int i = 0; i < 8; ++i) ls[i] = 1.f / ls[i];
#pragma unroll
  for (int n = 0; n < 4; ++n)
#pragma unroll
    for (int i = 0; i < 8; ++i) {
      int qg = qbase + w * 16 + i + hi * 8;
      obuf[(size_t)(b * TT + qg) * DD + h * 64 + n * 16 + r] = f2bf(o[n][i] * ls[i]);
    }
}

// ---------- launch ----------
extern "C" void kernel_launch(void* const* d_in, const int* in_sizes, int n_in,
                              void* d_out, int out_size, void* d_ws, size_t ws_size,
                              hipStream_t stream) {
  (void)in_sizes; (void)n_in; (void)out_size; (void)ws_size;
  const int*   x     = (const int*)  d_in[0];
  const float* tok   = (const float*)d_in[1];
  const float* pos   = (const float*)d_in[2];
  const float* attnw = (const float*)d_in[3];
  const float* attnb = (const float*)d_in[4];
  const float* projw = (const float*)d_in[5];
  const float* projb = (const float*)d_in[6];
  const float* ln1g  = (const float*)d_in[7];
  const float* ln1b  = (const float*)d_in[8];
  const float* ln2g  = (const float*)d_in[9];
  const float* ln2b  = (const float*)d_in[10];
  const float* ff1w  = (const float*)d_in[11];
  const float* ff1b  = (const float*)d_in[12];
  const float* ff2w  = (const float*)d_in[13];
  const float* ff2b  = (const float*)d_in[14];
  const float* lnfg  = (const float*)d_in[15];
  const float* lnfb  = (const float*)d_in[16];

  char* ws = (char*)d_ws; size_t off = 0;
  auto carve = [&](size_t bytes) { char* p = ws + off; off += (bytes + 255) & ~(size_t)255; return p; };
  float*  h    = (float*) carve((size_t)MTOK * DD * 4);        // residual stream f32
  __bf16* hn   = (__bf16*)carve((size_t)MTOK * DD * 2);        // LN outputs bf16
  __bf16* qkvb = (__bf16*)carve((size_t)MTOK * 3 * DD * 2);    // QKV bf16
  __bf16* ob   = (__bf16*)carve((size_t)MTOK * DD * 2);        // attention out bf16
  __bf16* ffb  = (__bf16*)carve((size_t)MTOK * FFD * 2);       // FF activation bf16
  // pre-converted bf16 [N,K] weights
  const size_t SZ_A = (size_t)3 * DD * DD;   // 1536x512
  const size_t SZ_P = (size_t)DD * DD;       // 512x512
  const size_t SZ_1 = (size_t)FFD * DD;      // 2048x512
  const size_t SZ_2 = (size_t)DD * FFD;      // 512x2048
  const size_t LW   = SZ_A + SZ_P + SZ_1 + SZ_2;
  __bf16* wts  = (__bf16*)carve((size_t)LLAY * LW * 2);
  __bf16* tokT = (__bf16*)carve((size_t)VV * DD * 2);

  // one-time weight transpose+convert passes (~150MB traffic, ~10us at HBM peak)
  cvt_kernel<<<(VV * DD) / 256, 256, 0, stream>>>(tok, tokT);
  for (int l = 0; l < LLAY; ++l) {
    __bf16* aT = wts + (size_t)l * LW;
    __bf16* pT = aT + SZ_A;
    __bf16* f1 = pT + SZ_P;
    __bf16* f2 = f1 + SZ_1;
    tcvt_kernel<<<dim3(3 * DD / 32, DD / 32),  256, 0, stream>>>(attnw + (size_t)l * DD * 3 * DD, aT, DD, 3 * DD);
    tcvt_kernel<<<dim3(DD / 32, DD / 32),      256, 0, stream>>>(projw + (size_t)l * DD * DD,     pT, DD, DD);
    tcvt_kernel<<<dim3(FFD / 32, DD / 32),     256, 0, stream>>>(ff1w  + (size_t)l * DD * FFD,    f1, DD, FFD);
    tcvt_kernel<<<dim3(DD / 32, FFD / 32),     256, 0, stream>>>(ff2w  + (size_t)l * FFD * DD,    f2, FFD, DD);
  }

  embed_kernel<<<MTOK * DD / 256, 256, 0, stream>>>(x, tok, pos, h);
  for (int l = 0; l < LLAY; ++l) {
    const __bf16* aT = wts + (size_t)l * LW;
    const __bf16* pT = aT + SZ_A;
    const __bf16* f1 = pT + SZ_P;
    const __bf16* f2 = f1 + SZ_1;
    ln_kernel<<<MTOK / 8, 256, 0, stream>>>(h, ln1g + l * DD, ln1b + l * DD, hn);
    gemm_kernel<true,false,false,true,false,true><<<dim3(12, 64), 256, 0, stream>>>(
        hn, aT, attnb + (size_t)l * 3 * DD, nullptr, nullptr, qkvb, 3 * DD, DD);
    attn_kernel<<<dim3(TT / 64, BB * HH), 128, 0, stream>>>(qkvb, ob);
    gemm_kernel<true,false,true,false,true,false><<<dim3(4, 64), 256, 0, stream>>>(
        ob, pT, projb + (size_t)l * DD, h, h, nullptr, DD, DD);
    ln_kernel<<<MTOK / 8, 256, 0, stream>>>(h, ln2g + l * DD, ln2b + l * DD, hn);
    gemm_kernel<true,true,false,true,false,false><<<dim3(16, 64), 256, 0, stream>>>(
        hn, f1, ff1b + (size_t)l * FFD, nullptr, nullptr, ffb, FFD, DD);
    gemm_kernel<true,false,true,false,true,false><<<dim3(4, 64), 256, 0, stream>>>(
        ffb, f2, ff2b + (size_t)l * DD, h, h, nullptr, DD, FFD);
  }
  ln_kernel<<<MTOK / 8, 256, 0, stream>>>(h, lnfg, lnfb, hn);
  gemm_kernel<false,false,false,false,true,false><<<dim3(VV / 128, 64), 256, 0, stream>>>(
      hn, tokT, nullptr, nullptr, (float*)d_out, nullptr, VV, DD);
}
